// RWKVBlock_4234837754304
// MI455X (gfx1250) — matrix operand order repaired and checked by reading, not compiled, not run on hardware
//
#include <hip/hip_runtime.h>
#include <hip/hip_bf16.h>

// ---------------------------------------------------------------------------
// RWKV block stack for MI455X (gfx1250, wave32, WMMA).
// All GEMMs run through v_wmma_f32_16x16x32_bf16 with fused pre/epilogues;
// the wkv recurrence is a dedicated sequential-scan kernel.
// ---------------------------------------------------------------------------

typedef __attribute__((ext_vector_type(16))) __bf16 v16bf;
typedef __attribute__((ext_vector_type(8)))  float  v8f;

union Frag16 {
  v16bf v;
  unsigned int u[8];
};

__device__ __forceinline__ unsigned int pack_bf2(float lo, float hi) {
  // f32 -> bf16 with round-half-up, packed {hi.bf16, lo.bf16}.
  // 2x v_add_u32 + 1x v_perm_b32; no 16-bit subregister traffic.
  const unsigned int lr = __float_as_uint(lo) + 0x8000u;
  const unsigned int hr = __float_as_uint(hi) + 0x8000u;
  return __builtin_amdgcn_perm(hr, lr, 0x07060302u);
}

// Tile sizes: block = 256 threads = 8 waves; each wave computes 32x32 of C
// (2 A-frags x 2 B-frags = 4 WMMAs per K-step).
#define BM 256
#define BN 32
#define BK 32

// PRE: 0 = none, 1 = tanh(x), 2 = x * preaux (elementwise, same shape as X)
// EPI: 0 = none, 1 = tanh, 2 = sigmoid, 3 = * epiaux, 4 = + epiaux
template <int PRE, int EPI, bool HAS_SCALE>
__global__ __launch_bounds__(256) void gemm_wmma_bf16(
    const float* __restrict__ X,       // (M, K) row-major
    const float* __restrict__ W,       // (O, K) row-major  (Y = X @ W^T)
    const float* __restrict__ scale,   // (K,) per-input-channel scale or null
    const float* __restrict__ preaux,  // (M, K) or null
    const float* __restrict__ epiaux,  // (M, O) or null
    float* __restrict__ Y,             // (M, O)
    int K, int O) {
  __shared__ unsigned short lx[BM][BK + 2];  // A tile, bf16 bits, [m][k]
  __shared__ unsigned short lw[BN][BK + 2];  // B tile, bf16 bits, [n][k]

  const int tid  = threadIdx.x;
  const int wid  = tid >> 5;         // wave id 0..7
  const int lane = tid & 31;
  const int hsel = lane >> 4;        // 0: lanes 0-15, 1: lanes 16-31
  const int l16  = lane & 15;
  const long mbase = (long)blockIdx.y * BM;
  const int  nbase = blockIdx.x * BN;
  const int  mrow  = wid * 32;       // wave's row offset inside block tile

  v8f acc[2][2];
#pragma unroll
  for (int mi = 0; mi < 2; ++mi)
#pragma unroll
    for (int ni = 0; ni < 2; ++ni)
      acc[mi][ni] = (v8f){0.f, 0.f, 0.f, 0.f, 0.f, 0.f, 0.f, 0.f};

  for (int kb = 0; kb < K; kb += BK) {
    // ---- stage X tile: float4 loads, fused pre-op + tm scale, bf16 pack ----
#pragma unroll
    for (int i = tid; i < BM * BK / 4; i += 256) {
      const int m  = i >> 3;          // 8 float4 per 32-wide row
      const int k4 = (i & 7) * 4;
      const long g = (mbase + m) * (long)K + kb + k4;
      float4 xv = *reinterpret_cast<const float4*>(&X[g]);
      if (PRE == 1) {
        xv.x = tanhf(xv.x); xv.y = tanhf(xv.y);
        xv.z = tanhf(xv.z); xv.w = tanhf(xv.w);
      }
      if (PRE == 2) {
        const float4 pv = *reinterpret_cast<const float4*>(&preaux[g]);
        xv.x *= pv.x; xv.y *= pv.y; xv.z *= pv.z; xv.w *= pv.w;
      }
      if (HAS_SCALE) {
        const float4 sv = *reinterpret_cast<const float4*>(&scale[kb + k4]);
        xv.x *= sv.x; xv.y *= sv.y; xv.z *= sv.z; xv.w *= sv.w;
      }
      *reinterpret_cast<unsigned int*>(&lx[m][k4])     = pack_bf2(xv.x, xv.y);
      *reinterpret_cast<unsigned int*>(&lx[m][k4 + 2]) = pack_bf2(xv.z, xv.w);
    }
    // ---- stage W tile (one float4 per thread) ----
    {
      const int n  = tid >> 3;
      const int k4 = (tid & 7) * 4;
      const float4 wv =
          *reinterpret_cast<const float4*>(&W[(long)(nbase + n) * K + kb + k4]);
      *reinterpret_cast<unsigned int*>(&lw[n][k4])     = pack_bf2(wv.x, wv.y);
      *reinterpret_cast<unsigned int*>(&lw[n][k4 + 2]) = pack_bf2(wv.z, wv.w);
    }
    // prefetch next X K-slab (emits global_prefetch_b8)
    if (kb + BK < K)
      __builtin_prefetch(&X[(mbase + tid) * (long)K + kb + BK], 0, 1);
    __syncthreads();

    // ---- build fragments per ISA 7.12.2 bf16 layouts ----
    Frag16 a[2], b[2];
#pragma unroll
    for (int f = 0; f < 2; ++f) {
      const int am = mrow + 16 * f + l16;  // A row = lane%16 within frag
#pragma unroll
      for (int p = 0; p < 8; ++p) {
        // A lane<16 holds K {0..7,16..23}; lane>=16 holds K {8..15,24..31}
        const int k0 = 2 * p + 8 * hsel + ((p >= 4) ? 8 : 0);
        a[f].u[p] = *reinterpret_cast<const unsigned int*>(&lx[am][k0]);
      }
    }
#pragma unroll
    for (int f = 0; f < 2; ++f) {
#pragma unroll
      for (int p = 0; p < 8; ++p) {
        // B: column = lane%16; lane<16 holds K {0..7,16..23}; lane>=16 holds K {8..15,24..31}
        const int kk = 2 * p + 8 * hsel + ((p >= 4) ? 8 : 0);
        b[f].u[p] =
            *reinterpret_cast<const unsigned int*>(&lw[16 * f + l16][kk]);
      }
    }
#pragma unroll
    for (int mi = 0; mi < 2; ++mi)
#pragma unroll
      for (int ni = 0; ni < 2; ++ni)
        acc[mi][ni] = __builtin_amdgcn_wmma_f32_16x16x32_bf16(
            false, a[mi].v, false, b[ni].v, (short)0, acc[mi][ni], false,
            false);
    __syncthreads();
  }

  // ---- epilogue + store (D layout: row = 8*hsel + j, col = lane%16) ----
#pragma unroll
  for (int mi = 0; mi < 2; ++mi) {
#pragma unroll
    for (int j = 0; j < 8; ++j) {
      const long m = mbase + mrow + 16 * mi + 8 * hsel + j;
#pragma unroll
      for (int ni = 0; ni < 2; ++ni) {
        const int n = nbase + 16 * ni + l16;
        float v = acc[mi][ni][j];
        if (EPI == 1) v = tanhf(v);
        if (EPI == 2) v = 1.0f / (1.0f + __expf(-v));
        if (EPI == 3) v *= epiaux[m * O + n];
        if (EPI == 4) v += epiaux[m * O + n];
        Y[m * O + n] = v;
      }
    }
  }
}

// ---------------------------------------------------------------------------
// wkv recurrence: one thread per (batch n, channel c); sequential over T.
// Lanes read k/v at consecutive c -> fully coalesced b32 loads each step.
// ---------------------------------------------------------------------------
__global__ __launch_bounds__(256) void wkv_scan_kernel(
    const float* __restrict__ k, const float* __restrict__ v,
    const float* __restrict__ decay, const float* __restrict__ first,
    float* __restrict__ out, int T, int C, int total) {
  const int idx = blockIdx.x * blockDim.x + threadIdx.x;
  if (idx >= total) return;
  const int c = idx % C;
  const int n = idx / C;
  const float w = __expf(decay[c]);
  const float u = first[c];
  float alpha = 0.f, beta = 0.f, eps = 0.f;
  long base = (long)n * T * C + c;
#pragma unroll 4
  for (int t = 0; t < T; ++t, base += C) {
    const float kt = k[base];
    const float vt = v[base];
    const float ukt = u + kt;
    const float tau = fmaxf(ukt, eps);
    const float e1 = __expf(eps - tau);
    const float e2 = __expf(ukt - tau);
    out[base] = (e1 * alpha + e2 * vt) / (e1 * beta + e2);
    const float weps = eps - w;
    const float epsn = fmaxf(weps, kt);
    const float e1n = __expf(weps - epsn);
    const float e2n = __expf(kt - epsn);
    alpha = e1n * alpha + e2n * vt;
    beta  = e1n * beta + e2n;
    eps   = epsn;
  }
}

__global__ __launch_bounds__(256) void tanh4_kernel(
    const float4* __restrict__ in, float4* __restrict__ out, long n4) {
  const long i = (long)blockIdx.x * blockDim.x + threadIdx.x;
  if (i < n4) {
    float4 v = in[i];
    v.x = tanhf(v.x); v.y = tanhf(v.y);
    v.z = tanhf(v.z); v.w = tanhf(v.w);
    out[i] = v;
  }
}

// ---------------------------------------------------------------------------
// Host side
// ---------------------------------------------------------------------------
static const long MTOT = 16L * 4096L;  // N * T token rows
static const int  TT = 4096, NB = 16;

static void launch_gemm(int pre, int epi, bool sc, const float* X,
                        const float* W, const float* scale,
                        const float* paux, const float* eaux, float* Y, int K,
                        int O, hipStream_t s) {
  dim3 grid(O / BN, (unsigned)(MTOT / BM));
  dim3 blk(256);
#define GEMM_CASE(P, E, S)                                                    \
  gemm_wmma_bf16<P, E, S><<<grid, blk, 0, s>>>(X, W, scale, paux, eaux, Y, K, O)
  if      (pre == 0 && epi == 0 && !sc) GEMM_CASE(0, 0, false);
  else if (pre == 0 && epi == 0 &&  sc) GEMM_CASE(0, 0, true);
  else if (pre == 0 && epi == 1 &&  sc) GEMM_CASE(0, 1, true);
  else if (pre == 0 && epi == 2 &&  sc) GEMM_CASE(0, 2, true);
  else if (pre == 0 && epi == 3 && !sc) GEMM_CASE(0, 3, false);
  else if (pre == 1 && epi == 1 &&  sc) GEMM_CASE(1, 1, true);
  else if (pre == 1 && epi == 2 &&  sc) GEMM_CASE(1, 2, true);
  else if (pre == 2 && epi == 4 && !sc) GEMM_CASE(2, 4, false);
#undef GEMM_CASE
}

extern "C" void kernel_launch(void* const* d_in, const int* in_sizes, int n_in,
                              void* d_out, int out_size, void* d_ws,
                              size_t ws_size, hipStream_t stream) {
  (void)in_sizes; (void)n_in; (void)out_size; (void)ws_size;
  // Input order: JAX pytree flattening (dicts in sorted-key order).
  //  0: x
  //  1: params.Wpred
  //  2+5i .. : channels[i]{Wk,Wr,Wv,tmk,tmr}    i=0..3
  //  22..26  : preffn{Wk,Wr,Wv,tmk,tmr}
  //  27+9i.. : times[i]{Wk,Wo,Wr,Wv,time_decay,time_first,tmk,tmr,tmv}
  //  63      : it (unused)
  const float* x_in  = (const float*)d_in[0];
  const float* Wpred = (const float*)d_in[1];
  const float *cWk[4], *cWr[4], *cWv[4], *ctmk[4], *ctmr[4];
  for (int i = 0; i < 4; ++i) {
    const int b = 2 + 5 * i;
    cWk[i]  = (const float*)d_in[b + 0];
    cWr[i]  = (const float*)d_in[b + 1];
    cWv[i]  = (const float*)d_in[b + 2];
    ctmk[i] = (const float*)d_in[b + 3];
    ctmr[i] = (const float*)d_in[b + 4];
  }
  const float* pWk  = (const float*)d_in[22];
  const float* pWr  = (const float*)d_in[23];
  const float* pWv  = (const float*)d_in[24];
  const float* ptmk = (const float*)d_in[25];
  const float* ptmr = (const float*)d_in[26];
  const float *tWk[4], *tWo[4], *tWr[4], *tWv[4], *tdec[4], *tfst[4],
              *ttmk[4], *ttmr[4], *ttmv[4];
  for (int i = 0; i < 4; ++i) {
    const int b = 27 + 9 * i;
    tWk[i]  = (const float*)d_in[b + 0];
    tWo[i]  = (const float*)d_in[b + 1];
    tWr[i]  = (const float*)d_in[b + 2];
    tWv[i]  = (const float*)d_in[b + 3];
    tdec[i] = (const float*)d_in[b + 4];
    tfst[i] = (const float*)d_in[b + 5];
    ttmk[i] = (const float*)d_in[b + 6];
    ttmr[i] = (const float*)d_in[b + 7];
    ttmv[i] = (const float*)d_in[b + 8];
  }

  // Workspace layout (floats)
  float* ws      = (float*)d_ws;
  float* bufBig  = ws;                    // M x 1024  (channel hidden / time K)
  float* bufR    = bufBig + MTOT * 1024;  // M x 320   (r / sigmoid(r))
  float* bufV    = bufR  + MTOT * 320;    // M x 320   (time V)
  float* bufWkv  = bufV  + MTOT * 320;    // M x 320
  float* bufXt   = bufWkv + MTOT * 320;   // M x 320   (tanh(x), reused as residual)
  float* bufX0   = bufXt + MTOT * 320;    // M x 320
  float* bufX1   = bufX0 + MTOT * 320;    // M x 320

  // ---- preffn: d=64, hidden=256, out=64 ----
  launch_gemm(0, 1, true,  x_in,  pWk, ptmk, nullptr, nullptr, bufBig, 64, 256, stream); // tanh(k)
  launch_gemm(0, 2, true,  x_in,  pWr, ptmr, nullptr, nullptr, bufR,   64,  64, stream); // sigmoid(r)
  launch_gemm(0, 3, false, bufBig, pWv, nullptr, nullptr, bufR, bufX0, 256,  64, stream); // r * (tanh(k)@Wv)

  // ---- 4 layers of time-mix + channel-mix ----
  for (int i = 0; i < 4; ++i) {
    const int d = 64 * (i + 1);
    const int h = 4 * d;
    const int cn = 64 * (i + 2);
    const long nd = MTOT * (long)d;
    const long n4 = nd / 4;

    // xt = tanh(x)  (used by 3 GEMMs and as the residual)
    tanh4_kernel<<<dim3((unsigned)((n4 + 255) / 256)), dim3(256), 0, stream>>>(
        (const float4*)bufX0, (float4*)bufXt, n4);

    launch_gemm(0, 0, true, bufXt, tWk[i], ttmk[i], nullptr, nullptr, bufBig, d, d, stream); // K
    launch_gemm(0, 0, true, bufXt, tWv[i], ttmv[i], nullptr, nullptr, bufV,   d, d, stream); // V
    launch_gemm(0, 2, true, bufXt, tWr[i], ttmr[i], nullptr, nullptr, bufR,   d, d, stream); // sigmoid(R)

    const int total = NB * d;
    wkv_scan_kernel<<<dim3((total + 255) / 256), dim3(256), 0, stream>>>(
        bufBig, bufV, tdec[i], tfst[i], bufWkv, TT, d, total);

    // x = Wo @ (wkv * sr) + tanh(x)
    launch_gemm(2, 4, false, bufWkv, tWo[i], nullptr, bufR, bufXt, bufX1, d, d, stream);

    // channel-mix on tanh(x_time): k = tanh(.), r = sigmoid(.), out = r*(tanh(k)@Wv)
    launch_gemm(1, 1, true,  bufX1, cWk[i], ctmk[i], nullptr, nullptr, bufBig, d, h,  stream);
    launch_gemm(1, 2, true,  bufX1, cWr[i], ctmr[i], nullptr, nullptr, bufR,   d, cn, stream);
    launch_gemm(0, 3, false, bufBig, cWv[i], nullptr, nullptr, bufR, bufX0,  h, cn, stream);
  }

  // ---- final projection: (M,320) @ Wpred^T -> (M,32) ----
  launch_gemm(0, 0, false, bufX0, Wpred, nullptr, nullptr, nullptr,
              (float*)d_out, 320, 32, stream);
}
